// GRUState_16449724745586
// MI455X (gfx1250) — matrix operand order repaired and checked by reading, not compiled, not run on hardware
//
#include <hip/hip_runtime.h>

typedef __attribute__((ext_vector_type(16))) _Float16 v16h;
typedef __attribute__((ext_vector_type(8)))  float    v8f;

#define NUM_ITEMS_C 100000
#define EDIM 100
#define HDIM 200
#define TT   200
#define THREEH 600
#define NT600 38        // ceil(600/16) padded to 608 cols
#define NCOLP 608
#define KC_E 4          // E padded to 128 = 4 * 32
#define KC_H 7          // H padded to 224 = 7 * 32
#define NT_OUT 6250     // 100000 / 16
#define NTPW 4          // n-tiles per wave in output GEMM

// ---------------------------------------------------------------------------
// Pack a row-major f32 matrix W (Kvalid x Nvalid, leading dim ldw) into f16
// WMMA B-fragment layout: [n_tile][k_chunk][lane][16 halfs].
// B layout (16-bit, 16x16x32): lane L holds column N = L%16; lanes 0-15 hold
// K = c*32 + {0..7, 16..23}, lanes 16-31 hold K = c*32 + {8..15, 24..31} (2 halfs per VGPR).
// ---------------------------------------------------------------------------
__global__ void pack_b_kernel(const float* __restrict__ W, _Float16* __restrict__ dst,
                              int Kvalid, int Nvalid, long ldw, long total, int chunks) {
  long stride = (long)gridDim.x * blockDim.x;
  for (long i = (long)blockIdx.x * blockDim.x + threadIdx.x; i < total; i += stride) {
    int  j    = (int)(i & 15);
    int  L    = (int)((i >> 4) & 31);
    long rest = i >> 9;
    int  c    = (int)(rest % chunks);
    long nt   = rest / chunks;
    int  col  = (int)(nt * 16 + (L & 15));
    int  k    = c * 32 + ((L >= 16) ? 8 : 0) + ((j < 8) ? j : j + 8);
    float v = (col < Nvalid && k < Kvalid) ? W[(long)k * ldw + col] : 0.f;
    dst[i] = (_Float16)v;
  }
}

// Inverse of the 16-bit A-fragment layout: given (m, kk) within a 16x32 chunk,
// where does it live? lane<16 holds K {0..7, 16..23}; lane>=16 holds {8..15, 24..31}.
__device__ __forceinline__ void a_slot(int m, int kk, int& L, int& j) {
  if      (kk < 8)  { L = m;      j = kk;      }
  else if (kk < 16) { L = m + 16; j = kk - 8;  }
  else if (kk < 24) { L = m;      j = kk - 8;  }
  else              { L = m + 16; j = kk - 16; }
}

// ---------------------------------------------------------------------------
// GRU: 64 blocks, each owns 16 batch rows for all T=200 steps. h lives in LDS.
// Per step:   state -> A-frag (f16)   |  x = state@Wi + bi  (4 k-chunks)
//             hh = h@Wh               |  gates -> new h -> A-frag for next step
// ---------------------------------------------------------------------------
__global__ __launch_bounds__(256) void gru_kernel(
    const int* __restrict__ q, const int* __restrict__ f,
    const float* __restrict__ E_items, const float* __restrict__ E_rewards,
    const float* __restrict__ bi, const float* __restrict__ bhn,
    const _Float16* __restrict__ WiP, const _Float16* __restrict__ WhP,
    _Float16* __restrict__ hTP) {
  __shared__ _Float16 sA[KC_E * 32 * 16];     // state, WMMA A layout
  __shared__ _Float16 sH[KC_H * 32 * 16];     // h,     WMMA A layout (f16)
  __shared__ _Float16 xbuf[16 * NCOLP];       // x projections  (row-major)
  __shared__ _Float16 hhbuf[16 * NCOLP];      // h projections  (row-major)
  __shared__ float    hF[16 * HDIM];          // h, f32 master copy

  const int tid  = threadIdx.x;
  const int lane = tid & 31;
  const int wave = tid >> 5;
  const int wg   = blockIdx.x;                // 16-row batch slice

  for (int i = tid; i < 16 * HDIM; i += 256) hF[i] = 0.f;
  for (int i = tid; i < KC_H * 512; i += 256) sH[i] = (_Float16)0.f;
  __syncthreads();

  for (int t = 0; t < TT; ++t) {
    // --- build state A-fragment: state[m][k] = E_items[q]*E_rewards[f] -------
    for (int i = tid; i < 16 * 128; i += 256) {
      int m = i >> 7;
      int k = i & 127;
      float v = 0.f;
      if (k < EDIM) {
        int row = wg * 16 + m;
        int qi = q[row * TT + t];
        int fi = f[row * TT + t];
        v = E_items[(long)qi * EDIM + k] * E_rewards[fi * EDIM + k];
      }
      int L, j;
      a_slot(m, k & 31, L, j);
      sA[((k >> 5) * 32 + L) * 16 + j] = (_Float16)v;
    }
    __syncthreads();

    // --- A fragments, loaded once per wave per step --------------------------
    v16h ax[KC_E];
#pragma unroll
    for (int c = 0; c < KC_E; ++c)
      ax[c] = *(const v16h*)(sA + (c * 32 + lane) * 16);
    v16h ah[KC_H];
#pragma unroll
    for (int c = 0; c < KC_H; ++c)
      ah[c] = *(const v16h*)(sH + (c * 32 + lane) * 16);

    // --- both GEMMs, 38 column tiles split across 8 waves --------------------
    for (int nt = wave; nt < NT600; nt += 8) {
      int col = nt * 16 + (lane & 15);
      float bias = (col < THREEH) ? bi[col] : 0.f;
      v8f accx;
#pragma unroll
      for (int r = 0; r < 8; ++r) accx[r] = bias;
#pragma unroll
      for (int c = 0; c < KC_E; ++c) {
        v16h b = *(const v16h*)(WiP + ((long)(nt * KC_E + c) * 32 + lane) * 16);
        accx = __builtin_amdgcn_wmma_f32_16x16x32_f16(false, ax[c], false, b,
                                                      (short)0, accx, false, false);
      }
      v8f acch = {};
#pragma unroll
      for (int c = 0; c < KC_H; ++c) {
        v16h b = *(const v16h*)(WhP + ((long)(nt * KC_H + c) * 32 + lane) * 16);
        acch = __builtin_amdgcn_wmma_f32_16x16x32_f16(false, ah[c], false, b,
                                                      (short)0, acch, false, false);
      }
      int row0 = (lane < 16) ? 0 : 8;   // f32 C layout: VGPR r -> row r / r+8
#pragma unroll
      for (int r = 0; r < 8; ++r) {
        xbuf [(row0 + r) * NCOLP + col] = (_Float16)accx[r];
        hhbuf[(row0 + r) * NCOLP + col] = (_Float16)acch[r];
      }
    }
    __syncthreads();

    // --- gate math + h update ------------------------------------------------
    for (int i = tid; i < 16 * HDIM; i += 256) {
      int m = i / HDIM;
      int j = i - m * HDIM;
      float xr = (float)xbuf [m * NCOLP + j];
      float xz = (float)xbuf [m * NCOLP + HDIM + j];
      float xn = (float)xbuf [m * NCOLP + 2 * HDIM + j];
      float hr = (float)hhbuf[m * NCOLP + j];
      float hz = (float)hhbuf[m * NCOLP + HDIM + j];
      float hn = (float)hhbuf[m * NCOLP + 2 * HDIM + j];
      float r = 1.f / (1.f + __expf(-(xr + hr)));
      float z = 1.f / (1.f + __expf(-(xz + hz)));
      float n = tanhf(xn + r * (hn + bhn[j]));
      float h = (1.f - z) * n + z * hF[i];
      hF[i] = h;
      int L, jj;
      a_slot(m, j & 31, L, jj);
      sH[((j >> 5) * 32 + L) * 16 + jj] = (_Float16)h;  // K pads stay zero
    }
    __syncthreads();
  }

  // dump hT A-fragments for the output GEMM
  for (int i = tid; i < KC_H * 512; i += 256)
    hTP[(long)wg * (KC_H * 512) + i] = sH[i];
}

// ---------------------------------------------------------------------------
// out = hT @ W_out + b_out : 1024 x 100000, K = 200 (padded 224).
// Each wave: 16x64 tile (A frags in registers, 4 column tiles x 7 WMMAs).
// ---------------------------------------------------------------------------
__global__ __launch_bounds__(256) void out_gemm_kernel(
    const _Float16* __restrict__ hTP, const _Float16* __restrict__ WoP,
    const float* __restrict__ b_out, float* __restrict__ out) {
  const int lane = threadIdx.x & 31;
  const int wave = threadIdx.x >> 5;
  const int mt   = blockIdx.y;

  v16h a[KC_H];
#pragma unroll
  for (int c = 0; c < KC_H; ++c)
    a[c] = *(const v16h*)(hTP + ((long)(mt * KC_H + c) * 32 + lane) * 16);

  const int nt0 = (blockIdx.x * 8 + wave) * NTPW;
  for (int u = 0; u < NTPW; ++u) {
    int nt = nt0 + u;
    if (nt >= NT_OUT) return;                 // wave-uniform: EXEC stays all-1s
    int col = nt * 16 + (lane & 15);
    float bias = b_out[col];
    v8f acc;
#pragma unroll
    for (int r = 0; r < 8; ++r) acc[r] = bias;
#pragma unroll
    for (int c = 0; c < KC_H; ++c) {
      v16h b = *(const v16h*)(WoP + ((long)(nt * KC_H + c) * 32 + lane) * 16);
      acc = __builtin_amdgcn_wmma_f32_16x16x32_f16(false, a[c], false, b,
                                                   (short)0, acc, false, false);
    }
    long row0 = (long)mt * 16 + ((lane < 16) ? 0 : 8);
#pragma unroll
    for (int r = 0; r < 8; ++r)
      out[(row0 + r) * (long)NUM_ITEMS_C + col] = acc[r];
  }
}

// ---------------------------------------------------------------------------
extern "C" void kernel_launch(void* const* d_in, const int* in_sizes, int n_in,
                              void* d_out, int out_size, void* d_ws, size_t ws_size,
                              hipStream_t stream) {
  const int*   q         = (const int*)  d_in[0];
  const int*   f         = (const int*)  d_in[1];
  const float* E_items   = (const float*)d_in[2];
  const float* E_rewards = (const float*)d_in[3];
  const float* Wi        = (const float*)d_in[4];
  const float* bi        = (const float*)d_in[5];
  const float* Wh        = (const float*)d_in[6];
  const float* bhn       = (const float*)d_in[7];
  const float* W_out     = (const float*)d_in[8];
  const float* b_out     = (const float*)d_in[9];
  float*       out       = (float*)d_out;

  // workspace layout (f16 elements): WiP | WhP | WoP | hTP  (~45.7 MB total)
  const long tWi = (long)NT600 * KC_E * 512;     //    77,824
  const long tWh = (long)NT600 * KC_H * 512;     //   136,192
  const long tWo = (long)NT_OUT * KC_H * 512;    // 22,400,000
  _Float16* WiP = (_Float16*)d_ws;
  _Float16* WhP = WiP + tWi;
  _Float16* WoP = WhP + tWh;
  _Float16* hTP = WoP + tWo;

  pack_b_kernel<<<512,  256, 0, stream>>>(Wi,    WiP, EDIM, THREEH, THREEH, tWi, KC_E);
  pack_b_kernel<<<512,  256, 0, stream>>>(Wh,    WhP, HDIM, THREEH, THREEH, tWh, KC_H);
  pack_b_kernel<<<8192, 256, 0, stream>>>(W_out, WoP, HDIM, NUM_ITEMS_C, NUM_ITEMS_C, tWo, KC_H);

  gru_kernel<<<64, 256, 0, stream>>>(q, f, E_items, E_rewards, bi, bhn, WiP, WhP, hTP);

  dim3 g((NT_OUT + 8 * NTPW - 1) / (8 * NTPW), 64);   // (196, 64)
  out_gemm_kernel<<<g, 256, 0, stream>>>(hTP, WoP, b_out, out);
}
